// kMaXTransformerLayer_24730421691070
// MI455X (gfx1250) — compile-verified, matrix operand order checked by reading, not run on hardware
//
#include <hip/hip_runtime.h>
#include <hip/hip_bf16.h>
#include <math.h>

// ---------------------------------------------------------------------------
// Types / helpers
// ---------------------------------------------------------------------------
typedef __attribute__((ext_vector_type(16))) __bf16 v16bf;
typedef __attribute__((ext_vector_type(8)))  float  v8f;

#define EPS_BN 1e-3f

__device__ __forceinline__ unsigned short f2bf(float f) {
  unsigned int u = __float_as_uint(f);
  u += 0x7fffu + ((u >> 16) & 1u);           // round-to-nearest-even
  return (unsigned short)(u >> 16);
}
__device__ __forceinline__ float bf2f(unsigned short s) {
  return __uint_as_float(((unsigned int)s) << 16);
}
__device__ __forceinline__ unsigned pack2bf(float a, float b) {
  return (unsigned)f2bf(a) | ((unsigned)f2bf(b) << 16);
}
__device__ __forceinline__ float gelu_erf(float x) {
  return 0.5f * x * (1.0f + erff(x * 0.70710678118654752f));
}

// ---------------------------------------------------------------------------
// WMMA GEMM, token-major activations.
//   X   : bf16, [n][m][K]   (K contiguous per token)
//   W   : f32, arbitrary strides (wStrideO, wStrideK); may be per-batch
//   out : out[n,o,m] = act( BN( sum_k W[o,k]*X[n,m,k] + cbias[o] ) )
//   outF   f32  : m-major [n][m][O], or o-major [n][o][Mld] if oMajorF
//   outB   bf16 : m-major [n][m][O]
//   outBC  bf16 : c-major [n][o][Mld] (extra copy for stencil consumers)
// Block: 256 thr = 8 waves; tile 32(o) x 256(m); K-step 32.
// B tile staged with async global->LDS b128 copies (ASYNCcnt).
// Fragments are two contiguous 16B runs along K -> ds_load_b128 pairs.
// ---------------------------------------------------------------------------
#define MTILE 256
#define OTILE 32
#define KSTEP 32

__global__ __launch_bounds__(256)
void k_gemm2(const unsigned short* __restrict__ X,
             const float* __restrict__ W,
             const float* __restrict__ gamma,
             const float* __restrict__ beta,
             const float* __restrict__ cbias,
             float* __restrict__ outF,
             unsigned short* __restrict__ outB,
             unsigned short* __restrict__ outBC,
             int K, int O, int Mld, int Mvalid,
             long xBatch, long wBatch, long oBatchF, long oBatchB,
             int wStrideO, int wStrideK, int gbStride, int doGelu, int oMajorF)
{
  __shared__ unsigned short aT[OTILE * KSTEP];   // [o][k]  2 KB
  __shared__ unsigned short bT[MTILE * KSTEP];   // [m][k] 16 KB

  const int t  = threadIdx.x;
  const int m0 = blockIdx.x * MTILE;
  const int o0 = blockIdx.y * OTILE;
  const int n  = blockIdx.z;

  const unsigned short* Xn = X + (long)n * xBatch;
  const float*          Wn = W + (long)n * wBatch;

  v8f acc[4];
  #pragma unroll
  for (int j = 0; j < 4; ++j)
    #pragma unroll
    for (int r = 0; r < 8; ++r) acc[j][r] = 0.0f;

  const int wave = t >> 5;             // wave32
  const int lane = t & 31;
  const int g    = lane >> 4;          // lane half
  const int rn   = lane & 15;          // row (A) / col (B,D)
  const int wo   = wave >> 2;          // o sub-tile (0..1)
  const int wm   = wave & 3;           // m strip (0..3, 64 wide)

  const int ms = m0 + t;               // staging column owned by this thread
  const bool mOk = (ms < Mvalid);
  const unsigned ldsB = (unsigned)(unsigned long long)(&bT[t * KSTEP]);

  for (int kk = 0; kk < K; kk += KSTEP) {
    // ---- stage A tile (32 o x 32 k), f32 -> bf16 ----
    {
      const int ol = t >> 3;
      const int kq = (t & 7) * 4;
      const float* wp = Wn + (long)(o0 + ol) * wStrideO + (long)(kk + kq) * wStrideK;
      uint2 pk;
      if (wStrideK == 1) {
        const float4 wv = *(const float4*)wp;
        pk.x = pack2bf(wv.x, wv.y);
        pk.y = pack2bf(wv.z, wv.w);
        if (kk + KSTEP < K) __builtin_prefetch(wp + KSTEP, 0, 3);
      } else {
        pk.x = pack2bf(wp[0], wp[(long)wStrideK]);
        pk.y = pack2bf(wp[(long)2 * wStrideK], wp[(long)3 * wStrideK]);
      }
      *(uint2*)&aT[ol * KSTEP + kq] = pk;
    }
    // ---- stage B tile (256 m x 32 k): async global->LDS, 64B per thread ----
    if (mOk) {
      const unsigned long long ga =
          (unsigned long long)(const void*)(Xn + (long)ms * K + kk);
      asm volatile(
          "global_load_async_to_lds_b128 %0, %1, off\n\t"
          "global_load_async_to_lds_b128 %0, %1, off offset:16\n\t"
          "global_load_async_to_lds_b128 %0, %1, off offset:32\n\t"
          "global_load_async_to_lds_b128 %0, %1, off offset:48"
          :: "v"(ldsB), "v"(ga) : "memory");
    } else {
      uint4 z; z.x = z.y = z.z = z.w = 0u;
      uint4* d4 = (uint4*)&bT[t * KSTEP];
      d4[0] = z; d4[1] = z; d4[2] = z; d4[3] = z;
    }
    asm volatile("s_wait_asynccnt 0x0" ::: "memory");
    __syncthreads();

    // ---- fragments: two contiguous 16B runs along K ----
    union U { v16bf v; uint4 q[2]; };
    U af;
    {
      const unsigned short* ar = &aT[(wo * 16 + rn) * KSTEP + 8 * g];
      af.q[0] = *(const uint4*)ar;
      af.q[1] = *(const uint4*)(ar + 16);
    }
    #pragma unroll
    for (int j = 0; j < 4; ++j) {
      const int c = wm * 64 + j * 16 + rn;
      U bf;
      const unsigned short* br = &bT[c * KSTEP + 8 * g];
      bf.q[0] = *(const uint4*)br;
      bf.q[1] = *(const uint4*)(br + 16);
      acc[j] = __builtin_amdgcn_wmma_f32_16x16x32_bf16(
                   false, af.v, false, bf.v, (short)0, acc[j], false, false);
    }
    __syncthreads();
  }

  // ---- epilogue ----
  const float invs  = rsqrtf(1.0f + EPS_BN);
  const int   obase = o0 + wo * 16 + 8 * g;    // 8 consecutive o per lane
  float sc[8], bi[8], cb[8];
  #pragma unroll
  for (int r = 0; r < 8; ++r) {
    const int o = obase + r;
    sc[r] = gamma[o * gbStride] * invs;
    bi[r] = beta[o * gbStride];
    cb[r] = cbias ? cbias[o] : 0.0f;
  }
  #pragma unroll
  for (int j = 0; j < 4; ++j) {
    const int m = m0 + wm * 64 + j * 16 + rn;
    if (m >= Mvalid) continue;
    float v[8];
    #pragma unroll
    for (int r = 0; r < 8; ++r) {
      float x = (acc[j][r] + cb[r]) * sc[r] + bi[r];
      v[r] = doGelu ? gelu_erf(x) : x;
    }
    if (outF) {
      if (oMajorF) {
        #pragma unroll
        for (int r = 0; r < 8; ++r)
          outF[(long)n * oBatchF + (long)(obase + r) * Mld + m] = v[r];
      } else {
        float4 f0, f1;
        f0.x = v[0]; f0.y = v[1]; f0.z = v[2]; f0.w = v[3];
        f1.x = v[4]; f1.y = v[5]; f1.z = v[6]; f1.w = v[7];
        float* p = &outF[(long)n * oBatchF + (long)m * O + obase];
        *(float4*)p = f0; *(float4*)(p + 4) = f1;
      }
    }
    if (outB) {
      uint4 pk;
      pk.x = pack2bf(v[0], v[1]); pk.y = pack2bf(v[2], v[3]);
      pk.z = pack2bf(v[4], v[5]); pk.w = pack2bf(v[6], v[7]);
      *(uint4*)&outB[(long)n * oBatchB + (long)m * O + obase] = pk;
    }
    if (outBC) {
      #pragma unroll
      for (int r = 0; r < 8; ++r)
        outBC[(long)n * oBatchB + (long)(obase + r) * Mld + m] = f2bf(v[r]);
    }
  }
}

// ---------------------------------------------------------------------------
// Transpose / elementwise helpers
// ---------------------------------------------------------------------------
// X0[n][m][512] = bf16(gelu(pixel[n][c][m]))
__global__ void k_geluT_pixel(const float* __restrict__ src,
                              unsigned short* __restrict__ dst, int M) {
  const int m = blockIdx.x * blockDim.x + threadIdx.x;
  const int n = blockIdx.y;
  if (m >= M) return;
  const float* s = src + (long)n * 512 * M + m;
  unsigned short* d = dst + ((long)n * M + m) * 512;
  for (int c = 0; c < 512; ++c) d[c] = f2bf(gelu_erf(s[(long)c * M]));
}
// QFb[n][l][256] = bf16(query[n][c][l])
__global__ void k_cvtT_query(const float* __restrict__ src,
                             unsigned short* __restrict__ dst) {
  const int l = threadIdx.x;       // 128
  const int n = blockIdx.x;
  const float* s = src + (long)n * 256 * 128 + l;
  unsigned short* d = dst + ((long)n * 128 + l) * 256;
  for (int c = 0; c < 256; ++c) d[c] = f2bf(s[c * 128]);
}
__global__ void k_cvt_bf16(const float* __restrict__ src,
                           unsigned short* __restrict__ dst, long count) {
  long i = (long)blockIdx.x * blockDim.x + threadIdx.x;
  if (i < count) dst[i] = f2bf(src[i]);
}
__global__ void k_zero_f32(float* __restrict__ dst, long count) {
  long i = (long)blockIdx.x * blockDim.x + threadIdx.x;
  if (i < count) dst[i] = 0.0f;
}

// ---------------------------------------------------------------------------
// Depthwise 5x5x5 conv (SAME) + BN + GELU; in c-major bf16, out m-major bf16
// ---------------------------------------------------------------------------
__global__ __launch_bounds__(256)
void k_dwconv(const unsigned short* __restrict__ PSc,
              const float* __restrict__ wdw,
              const float* __restrict__ gdw, const float* __restrict__ bdw,
              unsigned short* __restrict__ out, int M)
{
  __shared__ float wsm[125];
  const int c = blockIdx.y, n = blockIdx.z;
  if (threadIdx.x < 125) wsm[threadIdx.x] = wdw[c * 125 + threadIdx.x];
  __syncthreads();
  const int m = blockIdx.x * blockDim.x + threadIdx.x;
  if (m >= M) return;
  const int x = m / 576, rem = m % 576, y = rem / 24, z = rem % 24;
  const unsigned short* in = PSc + ((long)n * 256 + c) * M;
  float acc = 0.0f;
  int wi = 0;
  for (int dx = -2; dx <= 2; ++dx) {
    const int xx = x + dx; const bool okx = (0 <= xx && xx < 24);
    for (int dy = -2; dy <= 2; ++dy) {
      const int yy = y + dy; const bool oky = (0 <= yy && yy < 24);
      for (int dz = -2; dz <= 2; ++dz, ++wi) {
        const int zz = z + dz;
        if (okx && oky && 0 <= zz && zz < 24)
          acc += wsm[wi] * bf2f(in[(xx * 24 + yy) * 24 + zz]);
      }
    }
  }
  const float invs = rsqrtf(1.0f + EPS_BN);
  const float v = acc * gdw[c] * invs + bdw[c];
  out[((long)n * M + m) * 256 + c] = f2bf(gelu_erf(v));
}

// L2-normalize over 128 channels (token-major) -> bf16 token-major
__global__ void k_l2norm(const float* __restrict__ PS2,
                         unsigned short* __restrict__ PN, int M)
{
  const int m = blockIdx.x * blockDim.x + threadIdx.x;
  const int n = blockIdx.y;
  if (m >= M) return;
  const float* p = PS2 + ((long)n * M + m) * 128;
  float s = 0.0f;
  for (int c = 0; c < 128; ++c) { const float v = p[c]; s += v * v; }
  const float r = 1.0f / fmaxf(sqrtf(s), 1e-12f);
  unsigned short* q = PN + ((long)n * M + m) * 128;
  for (int c = 0; c < 128; ++c) q[c] = f2bf(p[c] * r);
}

// class logits: out[n,l,o] = sum_c QS[n,l,c]*wc[o,c] + bc[o] + void bias
__global__ void k_class(const float* __restrict__ QS,
                        const float* __restrict__ wc,
                        const float* __restrict__ bc,
                        float* __restrict__ out)
{
  const int idx = blockIdx.x * blockDim.x + threadIdx.x;
  if (idx >= 2 * 128 * 33) return;
  const int o = idx % 33, l = (idx / 33) % 128, n = idx / (33 * 128);
  const float* q = QS + ((long)n * 128 + l) * 256;
  float a = 0.0f;
  for (int c = 0; c < 256; ++c) a += q[c] * wc[o * 256 + c];
  a += bc[o];
  if (o == 32) a += logf(288.0f);   // log((33-1)*0.9/(1-0.9))
  out[idx] = a;
}

// per-pixel argmax over L=128 cluster logits (mask_logits is o-major [l][m])
__global__ void k_argmax(const float* __restrict__ ML, int* __restrict__ LB, int M)
{
  const int m = blockIdx.x * blockDim.x + threadIdx.x;
  const int n = blockIdx.y;
  if (m >= M) return;
  const float* p = ML + (long)n * 128 * M + m;
  float best = p[0]; int bi = 0;
  for (int l = 1; l < 128; ++l) {
    const float v = p[(long)l * M];
    if (v > best) { best = v; bi = l; }
  }
  LB[n * M + m] = bi;
}

// scatter-add token-major pixel_value rows into cluster accumulators [l][256]
__global__ void k_scatter(const float* __restrict__ PV, const int* __restrict__ LB,
                          float* __restrict__ KM, int M)
{
  const int m = blockIdx.x * blockDim.x + threadIdx.x;
  const int n = blockIdx.y;
  if (m >= M) return;
  const int lb = LB[n * M + m];
  const float* pv = PV + ((long)n * M + m) * 256;
  float* km = KM + ((long)n * 128 + lb) * 256;
  for (int d = 0; d < 256; ++d)
    atomicAdd(&km[d], pv[d]);
}

// 8-head self-attention over L=128 queries; QKV token-major [l][640]
__global__ __launch_bounds__(128)
void k_attn(const float* __restrict__ QKV,
            const float* __restrict__ gsim, const float* __restrict__ bsim,
            const float* __restrict__ gret, const float* __restrict__ bret,
            unsigned short* __restrict__ RETB)
{
  __shared__ float sq[16 * 128], sk[16 * 128], sv[32 * 128];
  const int h = blockIdx.x, n = blockIdx.y, l = threadIdx.x;
  const float* base = QKV + (long)n * 128 * 640;
  for (int r = 0; r < 16; ++r) {
    sq[r * 128 + l] = base[(long)l * 640 + (h * 16 + r)];
    sk[r * 128 + l] = base[(long)l * 640 + (128 + h * 16 + r)];
  }
  for (int r = 0; r < 32; ++r)
    sv[r * 128 + l] = base[(long)l * 640 + (256 + h * 32 + r)];
  __syncthreads();

  float ql[16];
  #pragma unroll
  for (int d = 0; d < 16; ++d) ql[d] = sq[d * 128 + l];
  const float invs = rsqrtf(1.0f + EPS_BN);
  const float ssc = gsim[h] * invs, sbi = bsim[h];

  float mx = -3.0e38f;
  for (int m = 0; m < 128; ++m) {
    float dt = 0.0f;
    #pragma unroll
    for (int d = 0; d < 16; ++d) dt += ql[d] * sk[d * 128 + m];
    mx = fmaxf(mx, dt * ssc + sbi);
  }
  float ret[32];
  #pragma unroll
  for (int d = 0; d < 32; ++d) ret[d] = 0.0f;
  float den = 0.0f;
  for (int m = 0; m < 128; ++m) {
    float dt = 0.0f;
    #pragma unroll
    for (int d = 0; d < 16; ++d) dt += ql[d] * sk[d * 128 + m];
    const float e = expf(dt * ssc + sbi - mx);
    den += e;
    #pragma unroll
    for (int d = 0; d < 32; ++d) ret[d] += e * sv[d * 128 + m];
  }
  const float rden = 1.0f / den;
  unsigned short* out = RETB + ((long)n * 128 + l) * 256 + h * 32;
  for (int d = 0; d < 32; ++d) {
    const int c = h * 32 + d;
    const float v = ret[d] * rden * gret[c] * invs + bret[c];
    out[d] = f2bf(gelu_erf(v));
  }
}

// qf2 = gelu(query(c-major) + KU + AU (token-major)) -> token-major f32+bf16
__global__ void k_qf2(const float* __restrict__ QF, const float* __restrict__ KU,
                      const float* __restrict__ AU,
                      float* __restrict__ QF2F, unsigned short* __restrict__ QF2B)
{
  const int l = threadIdx.x;  // 128
  const int n = blockIdx.x;
  const float* q = QF + (long)n * 256 * 128 + l;
  const long tm = ((long)n * 128 + l) * 256;
  for (int c = 0; c < 256; ++c) {
    const float v = gelu_erf(q[c * 128] + KU[tm + c] + AU[tm + c]);
    QF2F[tm + c] = v; QF2B[tm + c] = f2bf(v);
  }
}

// out_qf[n][c][l] = gelu(qf2 + ffn2)  (back to reference c-major layout)
__global__ void k_final(const float* __restrict__ QF2F, const float* __restrict__ F2,
                        float* __restrict__ out)
{
  const int l = threadIdx.x;  // 128
  const int n = blockIdx.x;
  const long tm = ((long)n * 128 + l) * 256;
  float* o = out + (long)n * 256 * 128 + l;
  for (int c = 0; c < 256; ++c)
    o[c * 128] = gelu_erf(QF2F[tm + c] + F2[tm + c]);
}

// ---------------------------------------------------------------------------
// Host launch sequence
// ---------------------------------------------------------------------------
extern "C" void kernel_launch(void* const* d_in, const int* in_sizes, int n_in,
                              void* d_out, int out_size, void* d_ws, size_t ws_size,
                              hipStream_t stream) {
  (void)in_sizes; (void)n_in; (void)out_size; (void)ws_size;
  const int N = 2, M = 13824, L = 128;

  const float* pixel = (const float*)d_in[0];
  const float* query = (const float*)d_in[1];
  const float* w_qc1 = (const float*)d_in[2];
  const float* g_qc1 = (const float*)d_in[3];
  const float* b_qc1 = (const float*)d_in[4];
  const float* w_pc1 = (const float*)d_in[5];
  const float* g_pc1 = (const float*)d_in[6];
  const float* b_pc1 = (const float*)d_in[7];
  const float* w_qkv = (const float*)d_in[8];
  const float* g_qkv = (const float*)d_in[9];
  const float* b_qkv = (const float*)d_in[10];
  const float* w_pv  = (const float*)d_in[11];
  const float* g_pv  = (const float*)d_in[12];
  const float* b_pv  = (const float*)d_in[13];
  const float* g_sim = (const float*)d_in[14];
  const float* b_sim = (const float*)d_in[15];
  const float* g_ret = (const float*)d_in[16];
  const float* b_ret = (const float*)d_in[17];
  const float* w_c3  = (const float*)d_in[18];
  const float* g_c3  = (const float*)d_in[19];
  const float* b_c3  = (const float*)d_in[20];
  const float* w_km  = (const float*)d_in[21];
  const float* g_km  = (const float*)d_in[22];
  const float* b_km  = (const float*)d_in[23];
  const float* w_f1  = (const float*)d_in[24];
  const float* g_f1  = (const float*)d_in[25];
  const float* b_f1  = (const float*)d_in[26];
  const float* w_f2  = (const float*)d_in[27];
  const float* g_f2  = (const float*)d_in[28];
  const float* b_f2  = (const float*)d_in[29];
  const float* w_dw  = (const float*)d_in[30];
  const float* g_dw  = (const float*)d_in[31];
  const float* b_dw  = (const float*)d_in[32];
  const float* w_ph1 = (const float*)d_in[33];
  const float* g_ph1 = (const float*)d_in[34];
  const float* b_ph1 = (const float*)d_in[35];
  const float* w_ph2 = (const float*)d_in[36];
  const float* bias_ph2 = (const float*)d_in[37];
  const float* g_ph2 = (const float*)d_in[38];
  const float* b_ph2 = (const float*)d_in[39];
  const float* w_mask = (const float*)d_in[40];
  const float* g_mask = (const float*)d_in[41];
  const float* b_mask = (const float*)d_in[42];
  const float* w_cls  = (const float*)d_in[43];
  const float* b_cls  = (const float*)d_in[44];
  const float* g_psm  = (const float*)d_in[45];
  const float* b_psm  = (const float*)d_in[46];

  char* ws = (char*)d_ws;
  size_t off = 0;
  auto alloc = [&](size_t bytes) -> void* {
    off = (off + 255) & ~(size_t)255;
    void* p = ws + off; off += bytes; return p;
  };

  // token-major bf16 activations unless noted
  unsigned short* X0  = (unsigned short*)alloc((size_t)N * M * 512 * 2);
  unsigned short* PS  = (unsigned short*)alloc((size_t)N * M * 256 * 2); // m-major
  unsigned short* PSc = (unsigned short*)alloc((size_t)N * 256 * M * 2); // c-major (dwconv)
  float*          PV  = (float*)         alloc((size_t)N * M * 256 * 4); // m-major
  // X0 dead after GEMM1 -> reuse for DW and PH1
  unsigned short* DW  = X0;
  unsigned short* PH1 = X0 + (size_t)N * M * 256;
  float*          PS2 = (float*)         alloc((size_t)N * M * 128 * 4);
  unsigned short* PN  = (unsigned short*)alloc((size_t)N * M * 128 * 2);
  unsigned short* QFb = (unsigned short*)alloc((size_t)N * L * 256 * 2);
  float*          QSf = (float*)         alloc((size_t)N * L * 256 * 4);
  unsigned short* QSb = (unsigned short*)alloc((size_t)N * L * 256 * 2);
  float*          QKV = (float*)         alloc((size_t)N * L * 640 * 4);
  float*          MK  = (float*)         alloc((size_t)N * L * 128 * 4);
  int*            LB  = (int*)           alloc((size_t)N * M * 4);
  float*          KM  = (float*)         alloc((size_t)N * L * 256 * 4);
  unsigned short* KMb = (unsigned short*)alloc((size_t)N * L * 256 * 2);
  float*          KU  = (float*)         alloc((size_t)N * L * 256 * 4);
  unsigned short* RETB= (unsigned short*)alloc((size_t)N * L * 256 * 2);
  float*          AU  = (float*)         alloc((size_t)N * L * 256 * 4);
  float*          QF2F= (float*)         alloc((size_t)N * L * 256 * 4);
  unsigned short* QF2B= (unsigned short*)alloc((size_t)N * L * 256 * 2);
  unsigned short* F1  = (unsigned short*)alloc((size_t)N * L * 2048 * 2);
  float*          F2  = (float*)         alloc((size_t)N * L * 256 * 4);

  float* out_qf  = (float*)d_out;                 // N*256*L (c-major)
  float* out_cls = out_qf + (size_t)N * 256 * L;  // N*L*33
  float* out_ml  = out_cls + (size_t)N * L * 33;  // N*L*M (o-major)

  const int MB = (M + MTILE - 1) / MTILE;         // 54
  const long QT = (long)L * 256;                  // query-side token-major stride

  // 1) X0 = bf16(gelu(pixel)), transposed to token-major
  k_geluT_pixel<<<dim3(MB, N), 256, 0, stream>>>(pixel, X0, M);
  // 2) pixel_space = gelu(bn(w_pc1 @ X0))  [512->256], m-major + c-major copies
  k_gemm2<<<dim3(MB, 8, N), 256, 0, stream>>>(
      X0, w_pc1, g_pc1, b_pc1, nullptr, nullptr, PS, PSc,
      512, 256, M, M, (long)M * 512, 0, 0, (long)M * 256, 512, 1, 1, 1, 0);
  // 3) pixel_value = bn(w_pv @ PS)  [256->256], f32 m-major
  k_gemm2<<<dim3(MB, 8, N), 256, 0, stream>>>(
      PS, w_pv, g_pv, b_pv, nullptr, PV, nullptr, nullptr,
      256, 256, M, M, (long)M * 256, 0, (long)M * 256, 0, 256, 1, 1, 0, 0);
  // 4) depthwise 5x5x5 + bn + gelu (c-major in, m-major out)
  k_dwconv<<<dim3(MB, 256, N), 256, 0, stream>>>(PSc, w_dw, g_dw, b_dw, DW, M);
  // 5) ph1 = gelu(bn(w_ph1 @ DW))  [256->256]
  k_gemm2<<<dim3(MB, 8, N), 256, 0, stream>>>(
      DW, w_ph1, g_ph1, b_ph1, nullptr, nullptr, PH1, nullptr,
      256, 256, M, M, (long)M * 256, 0, 0, (long)M * 256, 256, 1, 1, 1, 0);
  // 6) ps2 = bn(w_ph2 @ PH1 + bias_ph2)  [256->128], f32 m-major
  k_gemm2<<<dim3(MB, 4, N), 256, 0, stream>>>(
      PH1, w_ph2, g_ph2, b_ph2, bias_ph2, PS2, nullptr, nullptr,
      256, 128, M, M, (long)M * 256, 0, (long)M * 128, 0, 256, 1, 1, 0, 0);
  // 7) pixel_norm (token-major)
  k_l2norm<<<dim3(MB, N), 256, 0, stream>>>(PS2, PN, M);

  // 8) query path: transpose to token-major bf16
  k_cvtT_query<<<N, 128, 0, stream>>>(query, QFb);
  // 9) query_space = gelu(bn(w_qc1 @ QF))
  k_gemm2<<<dim3(1, 8, N), 256, 0, stream>>>(
      QFb, w_qc1, g_qc1, b_qc1, nullptr, QSf, QSb, nullptr,
      256, 256, L, L, QT, 0, QT, QT, 256, 1, 1, 1, 0);
  // 10) qkv = bn(w_qkv @ QS)  [256->640]
  k_gemm2<<<dim3(1, 20, N), 256, 0, stream>>>(
      QSb, w_qkv, g_qkv, b_qkv, nullptr, QKV, nullptr, nullptr,
      256, 640, L, L, QT, 0, (long)L * 640, 0, 256, 1, 1, 0, 0);
  // 11) mask_kernel = bn(w_mask @ QS)  [256->128], token-major [l][128]
  k_gemm2<<<dim3(1, 4, N), 256, 0, stream>>>(
      QSb, w_mask, g_mask, b_mask, nullptr, MK, nullptr, nullptr,
      256, 128, L, L, QT, 0, (long)L * 128, 0, 256, 1, 1, 0, 0);
  // 12) mask_logits[l,m] = psm*( sum_c MK[l,c]*PN[m,c] ) + b_psm, o-major out
  k_gemm2<<<dim3(MB, 4, N), 256, 0, stream>>>(
      PN, MK, g_psm, b_psm, nullptr, out_ml, nullptr, nullptr,
      128, 128, M, M, (long)M * 128, (long)L * 128, (long)128 * M, 0,
      128 /*wStrideO(l)*/, 1 /*wStrideK(c)*/, 0 /*scalar BN*/, 0, 1 /*o-major*/);
  // 13) class logits
  k_class<<<(2 * 128 * 33 + 127) / 128, 128, 0, stream>>>(QSf, w_cls, b_cls, out_cls);

  // 14-17) k-means hard assignment + cluster accumulation
  k_zero_f32<<<(N * L * 256 + 255) / 256, 256, 0, stream>>>(KM, (long)N * L * 256);
  k_argmax<<<dim3(MB, N), 256, 0, stream>>>(out_ml, LB, M);
  k_scatter<<<dim3(MB, N), 256, 0, stream>>>(PV, LB, KM, M);
  k_cvt_bf16<<<(N * L * 256 + 255) / 256, 256, 0, stream>>>(KM, KMb, (long)N * L * 256);
  // 18) kmeans_update = bn(w_km @ KM)
  k_gemm2<<<dim3(1, 8, N), 256, 0, stream>>>(
      KMb, w_km, g_km, b_km, nullptr, KU, nullptr, nullptr,
      256, 256, L, L, QT, 0, QT, 0, 256, 1, 1, 0, 0);
  // 19) self-attention -> gelu(bn(retrieved)) bf16 token-major
  k_attn<<<dim3(8, N), 128, 0, stream>>>(QKV, g_sim, b_sim, g_ret, b_ret, RETB);
  // 20) attn_update = bn(w_conv3 @ retrieved)
  k_gemm2<<<dim3(1, 8, N), 256, 0, stream>>>(
      RETB, w_c3, g_c3, b_c3, nullptr, AU, nullptr, nullptr,
      256, 256, L, L, QT, 0, QT, 0, 256, 1, 1, 0, 0);
  // 21) qf2 = gelu(query + KU + AU)
  k_qf2<<<N, 128, 0, stream>>>(query, KU, AU, QF2F, QF2B);
  // 22) ffn1 = gelu(bn(w_f1 @ qf2)) [256->2048]
  k_gemm2<<<dim3(1, 64, N), 256, 0, stream>>>(
      QF2B, w_f1, g_f1, b_f1, nullptr, nullptr, F1, nullptr,
      256, 2048, L, L, QT, 0, 0, (long)L * 2048, 256, 1, 1, 1, 0);
  // 23) ffn2 = bn(w_f2 @ f1) [2048->256]
  k_gemm2<<<dim3(1, 8, N), 256, 0, stream>>>(
      F1, w_f2, g_f2, b_f2, nullptr, F2, nullptr, nullptr,
      2048, 256, L, L, (long)L * 2048, 0, QT, 0, 2048, 1, 1, 0, 0);
  // 24) qf_out = gelu(qf2 + ffn2), back to c-major reference layout
  k_final<<<N, 128, 0, stream>>>(QF2F, F2, out_qf);
}
